// bwd_LSTM_146028888506
// MI455X (gfx1250) — matrix operand order repaired and checked by reading, not compiled, not run on hardware
//
#include <hip/hip_runtime.h>

// ---- constants for this problem ----
#define BB  4096   // batch
#define IDD 2048   // input dim
#define ODD 1024   // output (hidden) dim at level 1

// ---- WMMA vector types (per CDNA5_HIP.md probe-confirmed signatures) ----
typedef __attribute__((ext_vector_type(16))) __bf16 v16bf;
typedef __attribute__((ext_vector_type(8)))  float  v8f;

union AFrag { v16bf v; uint4 q[2]; };

__device__ __forceinline__ unsigned short f2bf(float f) {
    unsigned int u = __float_as_uint(f);
    unsigned int r = u + 0x7fffu + ((u >> 16) & 1u);   // round-to-nearest-even
    return (unsigned short)(r >> 16);
}

__device__ __forceinline__ float sigm(float x) { return 1.f / (1.f + __expf(-x)); }

// ---- fp32 -> bf16 conversion ----
__global__ void f32_to_bf16_kernel(const float* __restrict__ src,
                                   unsigned short* __restrict__ dst, int n) {
    int i = blockIdx.x * blockDim.x + threadIdx.x;
    if (i < n) dst[i] = f2bf(src[i]);
}

// ---- 4-gate x 2-row-tile WMMA accumulation over one K range ----
// aA/aB : per-lane A-row pointers for the two 16-row tiles (offset by (l/16)*8)
// w0..3 : per-lane W-row pointers for gates i,f,g,o (offset by (l/16)*8)
// Reuses each B (weight) fragment across both M tiles: 8 WMMA per 12 loads.
__device__ __forceinline__ void gemm_acc2(const unsigned short* aA, const unsigned short* aB,
                                          const unsigned short* w0, const unsigned short* w1,
                                          const unsigned short* w2, const unsigned short* w3,
                                          int K,
                                          v8f& p0, v8f& p1, v8f& p2, v8f& p3,
                                          v8f& q0, v8f& q1, v8f& q2, v8f& q3) {
#pragma unroll 2
    for (int k0 = 0; k0 < K; k0 += 32) {
        AFrag fa, fb, b0, b1, b2, b3;
        fa.q[0] = *(const uint4*)(aA + k0);          // K = half*8 + 0..7
        fa.q[1] = *(const uint4*)(aA + k0 + 16);     // K = 16 + half*8 + 0..7
        fb.q[0] = *(const uint4*)(aB + k0);
        fb.q[1] = *(const uint4*)(aB + k0 + 16);
        b0.q[0] = *(const uint4*)(w0 + k0); b0.q[1] = *(const uint4*)(w0 + k0 + 16);
        b1.q[0] = *(const uint4*)(w1 + k0); b1.q[1] = *(const uint4*)(w1 + k0 + 16);
        b2.q[0] = *(const uint4*)(w2 + k0); b2.q[1] = *(const uint4*)(w2 + k0 + 16);
        b3.q[0] = *(const uint4*)(w3 + k0); b3.q[1] = *(const uint4*)(w3 + k0 + 16);
        p0 = __builtin_amdgcn_wmma_f32_16x16x32_bf16(false, fa.v, false, b0.v, (short)0, p0, false, false);
        q0 = __builtin_amdgcn_wmma_f32_16x16x32_bf16(false, fb.v, false, b0.v, (short)0, q0, false, false);
        p1 = __builtin_amdgcn_wmma_f32_16x16x32_bf16(false, fa.v, false, b1.v, (short)0, p1, false, false);
        q1 = __builtin_amdgcn_wmma_f32_16x16x32_bf16(false, fb.v, false, b1.v, (short)0, q1, false, false);
        p2 = __builtin_amdgcn_wmma_f32_16x16x32_bf16(false, fa.v, false, b2.v, (short)0, p2, false, false);
        q2 = __builtin_amdgcn_wmma_f32_16x16x32_bf16(false, fb.v, false, b2.v, (short)0, q2, false, false);
        p3 = __builtin_amdgcn_wmma_f32_16x16x32_bf16(false, fa.v, false, b3.v, (short)0, p3, false, false);
        q3 = __builtin_amdgcn_wmma_f32_16x16x32_bf16(false, fb.v, false, b3.v, (short)0, q3, false, false);
    }
}

// ---- fused LSTM level: gates = X*Wih^T + H*Whh^T + b ; then cell math ----
// Each wave: 32 rows (2 M tiles) x 16 hidden cols x 4 gates.
// grid.x = H/16 (hidden col tiles), grid.y = M/256 (8 waves * 32 rows each)
__global__ __launch_bounds__(256)
void lstm_wmma_kernel(const unsigned short* __restrict__ Xb, int K1,
                      const unsigned short* __restrict__ Hb, int K2,
                      const unsigned short* __restrict__ Wih,
                      const unsigned short* __restrict__ Whh,
                      const float* __restrict__ bih, const float* __restrict__ bhh,
                      const float* __restrict__ Cin,
                      float* __restrict__ Hout, float* __restrict__ Cout, int H) {
    const int wave = threadIdx.x >> 5;
    const int lane = threadIdx.x & 31;
    const int half = lane >> 4;
    const int lr   = lane & 15;
    const int n0   = blockIdx.x << 4;
    const int m0   = (blockIdx.y << 8) + (wave << 5);   // 32 rows per wave
    const int col  = n0 + lr;

    v8f p0 = {}, p1 = {}, p2 = {}, p3 = {};   // rows m0 .. m0+15
    v8f q0 = {}, q1 = {}, q2 = {}, q3 = {};   // rows m0+16 .. m0+31

    // X * Wih^T
    gemm_acc2(Xb + (size_t)(m0 + lr) * K1 + half * 8,
              Xb + (size_t)(m0 + 16 + lr) * K1 + half * 8,
              Wih + ((size_t)0 * H + col) * K1 + half * 8,
              Wih + ((size_t)1 * H + col) * K1 + half * 8,
              Wih + ((size_t)2 * H + col) * K1 + half * 8,
              Wih + ((size_t)3 * H + col) * K1 + half * 8,
              K1, p0, p1, p2, p3, q0, q1, q2, q3);

    // H * Whh^T (level 4 passes Hb == nullptr: initial state is zero)
    if (Hb) {
        gemm_acc2(Hb + (size_t)(m0 + lr) * K2 + half * 8,
                  Hb + (size_t)(m0 + 16 + lr) * K2 + half * 8,
                  Whh + ((size_t)0 * H + col) * K2 + half * 8,
                  Whh + ((size_t)1 * H + col) * K2 + half * 8,
                  Whh + ((size_t)2 * H + col) * K2 + half * 8,
                  Whh + ((size_t)3 * H + col) * K2 + half * 8,
                  K2, p0, p1, p2, p3, q0, q1, q2, q3);
    }

    const float bi = bih[0 * H + col] + bhh[0 * H + col];
    const float bf = bih[1 * H + col] + bhh[1 * H + col];
    const float bg = bih[2 * H + col] + bhh[2 * H + col];
    const float bo = bih[3 * H + col] + bhh[3 * H + col];

#pragma unroll
    for (int t = 0; t < 2; ++t) {
        const v8f& a0 = t ? q0 : p0;
        const v8f& a1 = t ? q1 : p1;
        const v8f& a2 = t ? q2 : p2;
        const v8f& a3 = t ? q3 : p3;
#pragma unroll
        for (int v = 0; v < 8; ++v) {
            const int row = m0 + (t << 4) + (half << 3) + v;
            const size_t idx = (size_t)row * H + col;
            const float gi = a0[v] + bi;
            const float gf = a1[v] + bf;
            const float gg = a2[v] + bg;
            const float go = a3[v] + bo;
            const float cp = Cin ? Cin[idx] : 0.f;
            const float c2 = sigm(gf) * cp + sigm(gi) * tanhf(gg);
            const float h2 = sigm(go) * tanhf(c2);
            Hout[idx] = h2;
            Cout[idx] = c2;
        }
    }
}

// ---- conv1d(k=3,pad=1) + softmax over rows of length L (256/512/1024) ----
// one block per row; writes f32 (cell path) and/or bf16 (hidden->next GEMM)
__global__ __launch_bounds__(256)
void conv_softmax_kernel(const float* __restrict__ X, const float* __restrict__ w,
                         float* __restrict__ outF, unsigned short* __restrict__ outB, int L) {
    __shared__ float sx[1024];
    __shared__ float red[256];
    const int row = blockIdx.x;
    const int tid = threadIdx.x;
    const float* xr = X + (size_t)row * L;
    for (int j = tid; j < L; j += 256) sx[j] = xr[j];
    __syncthreads();
    const float w0 = w[0], w1 = w[1], w2 = w[2];
    const int cnt = L >> 8;                 // 1, 2 or 4
    float y[4];
    float m = -3.4e38f;
    for (int t = 0; t < cnt; ++t) {
        const int j = tid + (t << 8);
        const float xl = (j > 0)     ? sx[j - 1] : 0.f;
        const float xc = sx[j];
        const float xg = (j < L - 1) ? sx[j + 1] : 0.f;
        const float v  = w0 * xl + w1 * xc + w2 * xg;
        y[t] = v;
        m = fmaxf(m, v);
    }
    red[tid] = m; __syncthreads();
    for (int s = 128; s > 0; s >>= 1) { if (tid < s) red[tid] = fmaxf(red[tid], red[tid + s]); __syncthreads(); }
    m = red[0]; __syncthreads();
    float sum = 0.f;
    for (int t = 0; t < cnt; ++t) { y[t] = __expf(y[t] - m); sum += y[t]; }
    red[tid] = sum; __syncthreads();
    for (int s = 128; s > 0; s >>= 1) { if (tid < s) red[tid] += red[tid + s]; __syncthreads(); }
    const float inv = 1.f / red[0];
    for (int t = 0; t < cnt; ++t) {
        const int j = tid + (t << 8);
        const float r = y[t] * inv;
        if (outF) outF[(size_t)row * L + j] = r;
        if (outB) outB[(size_t)row * L + j] = f2bf(r);
    }
}

extern "C" void kernel_launch(void* const* d_in, const int* in_sizes, int n_in,
                              void* d_out, int out_size, void* d_ws, size_t ws_size,
                              hipStream_t stream) {
    (void)in_sizes; (void)n_in; (void)out_size; (void)ws_size;
    const float* spec = (const float*)d_in[0];
    const float* Wih1 = (const float*)d_in[1];  const float* Whh1 = (const float*)d_in[2];
    const float* bih1 = (const float*)d_in[3];  const float* bhh1 = (const float*)d_in[4];
    const float* Wih2 = (const float*)d_in[5];  const float* Whh2 = (const float*)d_in[6];
    const float* bih2 = (const float*)d_in[7];  const float* bhh2 = (const float*)d_in[8];
    const float* Wih3 = (const float*)d_in[9];  const float* Whh3 = (const float*)d_in[10];
    const float* bih3 = (const float*)d_in[11]; const float* bhh3 = (const float*)d_in[12];
    const float* Wih4 = (const float*)d_in[13]; const float* Whh4 = (const float*)d_in[14];
    const float* bih4 = (const float*)d_in[15]; const float* bhh4 = (const float*)d_in[16];
    const float* w2_1h = (const float*)d_in[17]; const float* w2_1c = (const float*)d_in[18];
    const float* w3_2h = (const float*)d_in[19]; const float* w3_2c = (const float*)d_in[20];
    const float* w4_3h = (const float*)d_in[21]; const float* w4_3c = (const float*)d_in[22];

    char* ws = (char*)d_ws;
    size_t off = 0;
    auto alloc = [&](size_t bytes) -> void* {
        void* p = ws + off;
        off = (off + bytes + 255) & ~(size_t)255;
        return p;
    };

    unsigned short* specb = (unsigned short*)alloc((size_t)BB * IDD * 2);
    unsigned short* W1i = (unsigned short*)alloc((size_t)4 * ODD * IDD * 2);
    unsigned short* W1h = (unsigned short*)alloc((size_t)4 * ODD * ODD * 2);
    unsigned short* W2i = (unsigned short*)alloc((size_t)2048 * 1024 * 2);
    unsigned short* W2h = (unsigned short*)alloc((size_t)2048 * 512 * 2);
    unsigned short* W3i = (unsigned short*)alloc((size_t)1024 * 512 * 2);
    unsigned short* W3h = (unsigned short*)alloc((size_t)1024 * 256 * 2);
    unsigned short* W4i = (unsigned short*)alloc((size_t)512 * 256 * 2);
    unsigned short* W4h = (unsigned short*)alloc((size_t)512 * 128 * 2);
    float*          hbuf = (float*)alloc((size_t)BB * ODD * 4);
    float*          cbuf = (float*)alloc((size_t)BB * ODD * 4);
    unsigned short* hbb  = (unsigned short*)alloc((size_t)BB * ODD * 2);
    float*          cbb  = (float*)alloc((size_t)BB * ODD * 4);

    auto cvt = [&](const float* s, unsigned short* d, size_t n) {
        f32_to_bf16_kernel<<<(unsigned)((n + 255) / 256), 256, 0, stream>>>(s, d, (int)n);
    };
    cvt(spec, specb, (size_t)BB * IDD);
    cvt(Wih1, W1i, (size_t)4 * ODD * IDD);
    cvt(Whh1, W1h, (size_t)4 * ODD * ODD);
    cvt(Wih2, W2i, (size_t)2048 * 1024);
    cvt(Whh2, W2h, (size_t)2048 * 512);
    cvt(Wih3, W3i, (size_t)1024 * 512);
    cvt(Whh3, W3h, (size_t)1024 * 256);
    cvt(Wih4, W4i, (size_t)512 * 256);
    cvt(Whh4, W4h, (size_t)512 * 128);

    const dim3 blk(256);

    // Level 4: M = B*8 = 32768, K1 = 256, H = 128, zero initial state
    lstm_wmma_kernel<<<dim3(128 / 16, 32768 / 256), blk, 0, stream>>>(
        specb, 256, nullptr, 0, W4i, nullptr, bih4, bhh4, nullptr, hbuf, cbuf, 128);
    conv_softmax_kernel<<<BB * 4, 256, 0, stream>>>(hbuf, w4_3h, nullptr, hbb, 256);
    conv_softmax_kernel<<<BB * 4, 256, 0, stream>>>(cbuf, w4_3c, cbb, nullptr, 256);

    // Level 3: M = B*4 = 16384, K1 = 512, K2 = 256, H = 256
    lstm_wmma_kernel<<<dim3(256 / 16, 16384 / 256), blk, 0, stream>>>(
        specb, 512, hbb, 256, W3i, W3h, bih3, bhh3, cbb, hbuf, cbuf, 256);
    conv_softmax_kernel<<<BB * 2, 256, 0, stream>>>(hbuf, w3_2h, nullptr, hbb, 512);
    conv_softmax_kernel<<<BB * 2, 256, 0, stream>>>(cbuf, w3_2c, cbb, nullptr, 512);

    // Level 2: M = B*2 = 8192, K1 = 1024, K2 = 512, H = 512
    lstm_wmma_kernel<<<dim3(512 / 16, 8192 / 256), blk, 0, stream>>>(
        specb, 1024, hbb, 512, W2i, W2h, bih2, bhh2, cbb, hbuf, cbuf, 512);
    conv_softmax_kernel<<<BB, 256, 0, stream>>>(hbuf, w2_1h, nullptr, hbb, 1024);
    conv_softmax_kernel<<<BB, 256, 0, stream>>>(cbuf, w2_1c, cbb, nullptr, 1024);

    // Level 1: M = B = 4096, K1 = 2048, K2 = 1024, H = 1024; h1 -> d_out
    lstm_wmma_kernel<<<dim3(1024 / 16, 4096 / 256), blk, 0, stream>>>(
        specb, 2048, hbb, 1024, W1i, W1h, bih1, bhh1, cbb, (float*)d_out, cbuf, 1024);
}
